// PathEncoder_76046690943401
// MI455X (gfx1250) — matrix operand order repaired and checked by reading, not compiled, not run on hardware
//
#include <hip/hip_runtime.h>

// ---------------------------------------------------------------------------
// MI455X (gfx1250, wave32) implementation of the PathEncoder forward pass.
// All dense GEMMs run on v_wmma_f32_16x16x32_f16 (f16 in, f32 accumulate).
// ---------------------------------------------------------------------------

typedef _Float16 v16h __attribute__((ext_vector_type(16)));
typedef _Float16 v8h  __attribute__((ext_vector_type(8)));
typedef float    v8f  __attribute__((ext_vector_type(8)));
typedef float    v4f  __attribute__((ext_vector_type(4)));

#define NNODES   32768
#define EDGES    524288
#define NGRAPH   256
#define SNODES   128
#define HIDD     256
#define FFND     512
#define G3       768          // 3 * HID (GRU gate width)
#define GLD      776          // gate LDS row stride (pad)
#define HLD      264          // state LDS row stride (pad)
#define NM       ((size_t)NNODES * HIDD)

// Anti-LICM device: an opaque SGPR zero, regenerated every loop iteration.
// Adding it to a pointer keeps the base kernel-arg provenance intact (so the
// loads stay global_load with SGPR base, not flat_load), while making the
// addresses non-loop-invariant so the GRU weight fragments are re-streamed
// from L2 each step instead of being hoisted and spilled to scratch.
__device__ __forceinline__ int opaque_zero() {
  int v = 0;
  asm volatile("" : "+s"(v));
  return v;
}

// ---------------------------------------------------------------------------
// WMMA fragment loads (wave32 layouts per CDNA5 ISA 7.12.2)
// A 16x32 f16: lane L -> row M = m0 + (L&15); hi = L>>4.
//   elems 0..7  : K = k0 + 8*hi + e           (contiguous 16B)
//   elems 8..15 : K = k0 + 16 + 8*hi + (e-8)  (contiguous 16B)
// B 32x16 f16 (from Wt[N][K] row-major): lane L -> col N = n0 + (L&15);
//   elems 0..7  : K = k0 + 8*hi + e           (contiguous 16B)
//   elems 8..15 : K = k0 + 16 + 8*hi + (e-8)  (contiguous 16B)
// C/D f32: lane L -> col N = n0+(L&15); VGPR j -> row M = m0 + j + 8*hi.
// ---------------------------------------------------------------------------

__device__ __forceinline__ v16h load_a_f32(const float* __restrict__ A, int lda,
                                           int m0, int k0) {
  const int lane = threadIdx.x & 31;
  const int m    = m0 + (lane & 15);
  const int hi   = lane >> 4;
  const float* p = A + (size_t)m * lda + k0 + 8 * hi;
  v4f c0 = *(const v4f*)(p + 0);
  v4f c1 = *(const v4f*)(p + 4);
  v4f c2 = *(const v4f*)(p + 16);
  v4f c3 = *(const v4f*)(p + 20);
  v16h r;
#pragma unroll
  for (int i = 0; i < 4; ++i) {
    r[i]      = (_Float16)c0[i];
    r[4 + i]  = (_Float16)c1[i];
    r[8 + i]  = (_Float16)c2[i];
    r[12 + i] = (_Float16)c3[i];
  }
  return r;
}

__device__ __forceinline__ v16h load_a_f16(const _Float16* __restrict__ A, int lda,
                                           int m0, int k0) {
  const int lane = threadIdx.x & 31;
  const int m    = m0 + (lane & 15);
  const int hi   = lane >> 4;
  const _Float16* p = A + (size_t)m * lda + k0 + 8 * hi;
  v8h c0 = *(const v8h*)(p);
  v8h c1 = *(const v8h*)(p + 16);
  v16h r;
#pragma unroll
  for (int i = 0; i < 8; ++i) { r[i] = c0[i]; r[8 + i] = c1[i]; }
  return r;
}

__device__ __forceinline__ v16h load_b_f16(const _Float16* __restrict__ W, int ldb,
                                           int n0, int k0) {
  const int lane = threadIdx.x & 31;
  const int n    = n0 + (lane & 15);
  const int hi   = lane >> 4;
  const _Float16* p = W + (size_t)n * ldb + k0 + 8 * hi;
  v8h c0 = *(const v8h*)(p);
  v8h c1 = *(const v8h*)(p + 16);
  v16h r;
#pragma unroll
  for (int i = 0; i < 8; ++i) { r[i] = c0[i]; r[8 + i] = c1[i]; }
  return r;
}

__device__ __forceinline__ v8f wmma16(v16h a, v16h b, v8f c) {
  return __builtin_amdgcn_wmma_f32_16x16x32_f16(false, a, false, b, (short)0, c,
                                                false, false);
}

// ---------------------------------------------------------------------------
// Generic GEMM:  C[M,N] = A[M,K] (f32) x Wt[N,K]^T (f16)  [+bias][+resid][relu]
// Block = 256 threads = 8 waves; BM=128 (wave w -> 16-row stripe), BN=64.
// FLAGS: 1=bias, 2=relu, 4=resid, 8=write f16 output
// ---------------------------------------------------------------------------
template <int FLAGS, int K>
__global__ void gemm_kernel(const float* __restrict__ A, int lda,
                            const _Float16* __restrict__ Wt, int ldb,
                            const float* __restrict__ bias,
                            const float* __restrict__ resid, int ldr,
                            float* __restrict__ C, _Float16* __restrict__ C16,
                            int ldc) {
  const int lane  = threadIdx.x & 31;
  const int wave  = threadIdx.x >> 5;
  const int m0    = blockIdx.x * 128 + wave * 16;
  const int nbase = blockIdx.y * 64;

  v8f acc[4] = {};
#pragma unroll 2
  for (int k0 = 0; k0 < K; k0 += 32) {
    v16h a = load_a_f32(A, lda, m0, k0);
#pragma unroll
    for (int ct = 0; ct < 4; ++ct) {
      v16h b = load_b_f16(Wt, ldb, nbase + ct * 16, k0);
      acc[ct] = wmma16(a, b, acc[ct]);
    }
  }
  const int hi = lane >> 4;
  const int nn = lane & 15;
#pragma unroll
  for (int ct = 0; ct < 4; ++ct) {
    const int n  = nbase + ct * 16 + nn;
    const float bv = (FLAGS & 1) ? bias[n] : 0.0f;
#pragma unroll
    for (int j = 0; j < 8; ++j) {
      const int m = m0 + j + 8 * hi;
      float v = acc[ct][j] + bv;
      if (FLAGS & 4) v += resid[(size_t)m * ldr + n];
      if (FLAGS & 2) v = fmaxf(v, 0.0f);
      if (FLAGS & 8) C16[(size_t)m * ldc + n] = (_Float16)v;
      else           C[(size_t)m * ldc + n]   = v;
    }
  }
}

// ---------------------------------------------------------------------------
// Fused 2-layer GRU over T=8 steps. One workgroup owns 16 nodes; h0/h1 state
// and [16 x 768] gate tiles live in LDS for the whole recurrence. Weights are
// re-read from (L2-hot) global memory every step via opaque offsets.
// ---------------------------------------------------------------------------
__device__ __forceinline__ void gru_gemm_to_lds(float* __restrict__ G,
                                                const float* __restrict__ A, int lda,
                                                int m0,
                                                const _Float16* __restrict__ Wt,
                                                int K) {
  const int lane = threadIdx.x & 31;
  const int wave = threadIdx.x >> 5;
  const int hi   = lane >> 4;
  const int nn   = lane & 15;
#pragma unroll
  for (int j = 0; j < 6; ++j) {          // 48 col tiles / 8 waves
    const int n0 = (wave + j * 8) * 16;
    v8f acc = {};
    for (int k0 = 0; k0 < K; k0 += 32) {
      v16h a = load_a_f32(A, lda, m0, k0);
      v16h b = load_b_f16(Wt, K, n0, k0);
      acc = wmma16(a, b, acc);
    }
#pragma unroll
    for (int jj = 0; jj < 8; ++jj)
      G[(jj + 8 * hi) * GLD + n0 + nn] = acc[jj];
  }
}

__device__ __forceinline__ void gru_update(float* __restrict__ h,
                                           const float* __restrict__ gi,
                                           const float* __restrict__ gh,
                                           const float* __restrict__ bih,
                                           const float* __restrict__ bhh) {
  for (int idx = threadIdx.x; idx < 16 * HIDD; idx += 256) {
    const int i = idx >> 8, j = idx & 255;
    const float ir = gi[i * GLD + j]             + bih[j];
    const float iz = gi[i * GLD + HIDD + j]      + bih[HIDD + j];
    const float in = gi[i * GLD + 2 * HIDD + j]  + bih[2 * HIDD + j];
    const float hr = gh[i * GLD + j]             + bhh[j];
    const float hz = gh[i * GLD + HIDD + j]      + bhh[HIDD + j];
    const float hn = gh[i * GLD + 2 * HIDD + j]  + bhh[2 * HIDD + j];
    const float r = 1.0f / (1.0f + __expf(-(ir + hr)));
    const float z = 1.0f / (1.0f + __expf(-(iz + hz)));
    const float n = tanhf(in + r * hn);
    h[i * HLD + j] = (1.0f - z) * n + z * h[i * HLD + j];
  }
}

__global__ void gru_kernel(const float* __restrict__ x,
                           const _Float16* __restrict__ Wih0,
                           const _Float16* __restrict__ Whh0,
                           const _Float16* __restrict__ Wih1,
                           const _Float16* __restrict__ Whh1,
                           const float* __restrict__ bih0,
                           const float* __restrict__ bhh0,
                           const float* __restrict__ bih1,
                           const float* __restrict__ bhh1,
                           float* __restrict__ hout) {
  extern __shared__ float sm[];
  float* gi = sm;                    // 16 x GLD
  float* gh = gi + 16 * GLD;         // 16 x GLD
  float* h0 = gh + 16 * GLD;         // 16 x HLD
  float* h1 = h0 + 16 * HLD;         // 16 x HLD
  const int node0 = blockIdx.x * 16;

  for (int i = threadIdx.x; i < 16 * HLD; i += 256) { h0[i] = 0.0f; h1[i] = 0.0f; }
  __syncthreads();

#pragma unroll 1
  for (int t = 0; t < 8; ++t) {
    // defeat LICM while keeping global-AS provenance: opaque zero offset is
    // regenerated each step, so weight/bias loads cannot be hoisted+spilled,
    // yet still lower to global_load (SGPR base), not flat_load.
    const int zz = opaque_zero();
    const _Float16* w_ih0 = Wih0 + zz;
    const _Float16* w_hh0 = Whh0 + zz;
    const _Float16* w_ih1 = Wih1 + zz;
    const _Float16* w_hh1 = Whh1 + zz;
    const float* p_bih0 = bih0 + zz;
    const float* p_bhh0 = bhh0 + zz;
    const float* p_bih1 = bih1 + zz;
    const float* p_bhh1 = bhh1 + zz;

    // layer 0: gi = x_t @ Wih0^T ; gh = h0 @ Whh0^T
    gru_gemm_to_lds(gi, x + t * 128, 1024, node0, w_ih0, 128);
    gru_gemm_to_lds(gh, h0, HLD, 0, w_hh0, 256);
    __syncthreads();
    gru_update(h0, gi, gh, p_bih0, p_bhh0);
    __syncthreads();
    // layer 1: gi = h0 @ Wih1^T ; gh = h1 @ Whh1^T
    gru_gemm_to_lds(gi, h0, HLD, 0, w_ih1, 256);
    gru_gemm_to_lds(gh, h1, HLD, 0, w_hh1, 256);
    __syncthreads();
    gru_update(h1, gi, gh, p_bih1, p_bhh1);
    __syncthreads();
  }
  for (int i = threadIdx.x; i < 16 * HIDD; i += 256) {
    const int r = i >> 8, c = i & 255;
    hout[(size_t)(node0 + r) * HIDD + c] = h1[r * HLD + c];
  }
}

// ---------------------------------------------------------------------------
// Attention: one workgroup per (graph, head). Q.K^T (WMMA), softmax, A.V (WMMA)
// ---------------------------------------------------------------------------
#define SLD 132   // f32 score row stride
#define ALD 136   // f16 A / Vt row stride

__global__ void attn_kernel(const _Float16* __restrict__ Q,
                            const _Float16* __restrict__ Kk,
                            const _Float16* __restrict__ V,
                            float* __restrict__ O) {
  extern __shared__ float sm[];
  float*     sS  = sm;                                 // 128 x SLD f32
  _Float16*  sA  = (_Float16*)(sS + 128 * SLD);        // 128 x ALD f16
  _Float16*  sVt = sA + 128 * ALD;                     //  32 x ALD f16
  const int b  = blockIdx.x >> 3;
  const int hh = blockIdx.x & 7;
  const size_t base = ((size_t)b * SNODES) * HIDD + hh * 32;
  const _Float16* Qg = Q  + base;
  const _Float16* Kg = Kk + base;
  const _Float16* Vg = V  + base;
  const int tid = threadIdx.x, lane = tid & 31, wave = tid >> 5;
  const int hi = lane >> 4, nn = lane & 15;

  // stage V transposed: sVt[d][k] = V[k][d]
  for (int i = tid; i < 32 * SNODES; i += 256) {
    const int d = i >> 7, k = i & 127;
    sVt[d * ALD + k] = Vg[(size_t)k * HIDD + d];
  }
  // scores = Q K^T / sqrt(32) ; wave -> row tile, 8 col tiles, K = 32 (1 WMMA)
  {
    const float scale = 0.17677669529663687f;
    v16h a = load_a_f16(Qg, HIDD, wave * 16, 0);
#pragma unroll
    for (int ct = 0; ct < 8; ++ct) {
      v16h bf = load_b_f16(Kg, HIDD, ct * 16, 0);
      v8f acc = {};
      acc = wmma16(a, bf, acc);
#pragma unroll
      for (int j = 0; j < 8; ++j)
        sS[(wave * 16 + j + 8 * hi) * SLD + ct * 16 + nn] = acc[j] * scale;
    }
  }
  __syncthreads();
  // row softmax -> f16 A
  if (tid < SNODES) {
    float mx = -1e30f;
    for (int k = 0; k < SNODES; ++k) mx = fmaxf(mx, sS[tid * SLD + k]);
    float s = 0.0f;
    for (int k = 0; k < SNODES; ++k) {
      const float e = __expf(sS[tid * SLD + k] - mx);
      sS[tid * SLD + k] = e; s += e;
    }
    const float inv = 1.0f / s;
    for (int k = 0; k < SNODES; ++k)
      sA[tid * ALD + k] = (_Float16)(sS[tid * SLD + k] * inv);
  }
  __syncthreads();
  // O = A V : wave -> row tile, 2 col tiles (dh=32), K = 128
#pragma unroll
  for (int ct = 0; ct < 2; ++ct) {
    v8f acc = {};
    for (int k0 = 0; k0 < SNODES; k0 += 32) {
      v16h a = load_a_f16(sA, ALD, wave * 16, k0);
      v16h bf = load_b_f16(sVt, ALD, ct * 16, k0);
      acc = wmma16(a, bf, acc);
    }
#pragma unroll
    for (int j = 0; j < 8; ++j) {
      const int q = wave * 16 + j + 8 * hi;
      O[((size_t)b * SNODES + q) * HIDD + hh * 32 + ct * 16 + nn] = acc[j];
    }
  }
}

// ---------------------------------------------------------------------------
// Small elementwise / reduction kernels
// ---------------------------------------------------------------------------
__global__ void cvt_kernel(const float* __restrict__ s, _Float16* __restrict__ d,
                           int rows, int cols, int trans) {
  const int i = blockIdx.x * 256 + threadIdx.x;
  if (i >= rows * cols) return;
  if (trans) {
    const int r = i / cols, c = i % cols;
    d[(size_t)c * rows + r] = (_Float16)s[i];
  } else {
    d[i] = (_Float16)s[i];
  }
}

__global__ void deg_init_kernel(float* __restrict__ deg) {
  const int i = blockIdx.x * 256 + threadIdx.x;
  if (i < NNODES) deg[i] = 1.0f;              // self loop
}

__global__ void deg_edge_kernel(const int* __restrict__ ei, float* __restrict__ deg) {
  const int e = blockIdx.x * 256 + threadIdx.x;
  if (e < EDGES) atomicAdd(&deg[ei[EDGES + e]], 1.0f);
}

__global__ void dinv_kernel(const float* __restrict__ deg, float* __restrict__ dinv) {
  const int i = blockIdx.x * 256 + threadIdx.x;
  if (i < NNODES) dinv[i] = rsqrtf(fmaxf(deg[i], 1e-12f));
}

__global__ void agg_init_kernel(const float* __restrict__ hw,
                                const float* __restrict__ dinv,
                                const float* __restrict__ bias,
                                float* __restrict__ agg) {
  const size_t i = (size_t)blockIdx.x * 256 + threadIdx.x;
  if (i >= NM) return;
  const int n = (int)(i >> 8), c = (int)(i & 255);
  const float dd = dinv[n];
  agg[i] = bias[c] + hw[i] * dd * dd;
}

__global__ void gcn_edge_kernel(const int* __restrict__ ei,
                                const float* __restrict__ hw,
                                const float* __restrict__ dinv,
                                float* __restrict__ agg) {
  const int idx = blockIdx.x * 256 + threadIdx.x;   // EDGES*64 threads
  const int e = idx >> 6;
  const int c = (idx & 63) << 2;
  const int s = ei[e];
  const int d = ei[EDGES + e];
  const float w = dinv[s] * dinv[d];
  const float* hp = hw + (size_t)s * HIDD + c;
  float* ap = agg + (size_t)d * HIDD + c;
  atomicAdd(ap + 0, hp[0] * w);
  atomicAdd(ap + 1, hp[1] * w);
  atomicAdd(ap + 2, hp[2] * w);
  atomicAdd(ap + 3, hp[3] * w);
}

__global__ void relu_kernel(const float* __restrict__ a, float* __restrict__ o) {
  const size_t i = (size_t)blockIdx.x * 256 + threadIdx.x;
  if (i < NM) o[i] = fmaxf(a[i], 0.0f);
}

__global__ void ln_kernel(const float* __restrict__ X, const float* __restrict__ g,
                          const float* __restrict__ bta, float* __restrict__ Y) {
  const int row  = blockIdx.x * 8 + (threadIdx.x >> 5);
  const int lane = threadIdx.x & 31;
  const float* xr = X + (size_t)row * HIDD;
  float v[8], s = 0.0f;
#pragma unroll
  for (int i = 0; i < 8; ++i) { v[i] = xr[lane + i * 32]; s += v[i]; }
#pragma unroll
  for (int off = 16; off; off >>= 1) s += __shfl_xor(s, off, 32);
  const float mu = s * (1.0f / 256.0f);
  float q = 0.0f;
#pragma unroll
  for (int i = 0; i < 8; ++i) { const float d = v[i] - mu; q += d * d; }
#pragma unroll
  for (int off = 16; off; off >>= 1) q += __shfl_xor(q, off, 32);
  const float inv = rsqrtf(q * (1.0f / 256.0f) + 1e-5f);
  float* yr = Y + (size_t)row * HIDD;
#pragma unroll
  for (int i = 0; i < 8; ++i) {
    const int c = lane + i * 32;
    yr[c] = (v[i] - mu) * inv * g[c] + bta[c];
  }
}

__global__ void pool_kernel(const float* __restrict__ h, float* __restrict__ out) {
  const int g = blockIdx.x, c = threadIdx.x;
  float s = 0.0f;
  for (int n = 0; n < SNODES; ++n) s += h[((size_t)g * SNODES + n) * HIDD + c];
  out[g * HIDD + c] = s * (1.0f / (float)SNODES);
}

// ---------------------------------------------------------------------------
// Host orchestration
// ---------------------------------------------------------------------------
static inline void launch_gemm(int flags, const float* A, int lda,
                               const _Float16* Wt, int K, int Ncols,
                               const float* bias, const float* resid, int ldr,
                               float* C, _Float16* C16, int ldc,
                               hipStream_t stream) {
  dim3 grid(NNODES / 128, Ncols / 64), block(256);
  if (K == 256) {
    switch (flags) {
      case 0: gemm_kernel<0, 256><<<grid, block, 0, stream>>>(A, lda, Wt, K, bias, resid, ldr, C, C16, ldc); break;
      case 3: gemm_kernel<3, 256><<<grid, block, 0, stream>>>(A, lda, Wt, K, bias, resid, ldr, C, C16, ldc); break;
      case 5: gemm_kernel<5, 256><<<grid, block, 0, stream>>>(A, lda, Wt, K, bias, resid, ldr, C, C16, ldc); break;
      case 9: gemm_kernel<9, 256><<<grid, block, 0, stream>>>(A, lda, Wt, K, bias, resid, ldr, C, C16, ldc); break;
    }
  } else {  // K == 512
    gemm_kernel<5, 512><<<grid, block, 0, stream>>>(A, lda, Wt, K, bias, resid, ldr, C, C16, ldc);
  }
}

extern "C" void kernel_launch(void* const* d_in, const int* in_sizes, int n_in,
                              void* d_out, int out_size, void* d_ws, size_t ws_size,
                              hipStream_t stream) {
  (void)in_sizes; (void)n_in; (void)out_size; (void)ws_size;

  const float* x  = (const float*)d_in[0];
  const int*   ei = (const int*)d_in[1];
  // d_in[2] = batch (graph id derivable, unused)
  const float* f_wih0 = (const float*)d_in[3];
  const float* f_whh0 = (const float*)d_in[4];
  const float* b_ih0  = (const float*)d_in[5];
  const float* b_hh0  = (const float*)d_in[6];
  const float* f_wih1 = (const float*)d_in[7];
  const float* f_whh1 = (const float*)d_in[8];
  const float* b_ih1  = (const float*)d_in[9];
  const float* b_hh1  = (const float*)d_in[10];
  // per-layer params: base 11 + l*18
  // 0:gcn_w 1:gcn_b 2:wq 3:bq 4:wk 5:bk 6:wv 7:bv 8:wo 9:bo
  // 10:ln0_g 11:ln0_b 12:ffn_w1 13:ffn_b1 14:ffn_w2 15:ffn_b2 16:ln1_g 17:ln1_b
  auto P = [&](int l, int j) { return (const float*)d_in[11 + l * 18 + j]; };

  // --- workspace carve (256B aligned) ---
  char* ws = (char*)d_ws;
  size_t off = 0;
  auto alloc = [&](size_t bytes) -> void* {
    size_t o = (off + 255) & ~(size_t)255;
    off = o + bytes;
    return (void*)(ws + o);
  };

  _Float16* Wih0 = (_Float16*)alloc((size_t)768 * 128 * 2);
  _Float16* Whh0 = (_Float16*)alloc((size_t)768 * 256 * 2);
  _Float16* Wih1 = (_Float16*)alloc((size_t)768 * 256 * 2);
  _Float16* Whh1 = (_Float16*)alloc((size_t)768 * 256 * 2);
  _Float16 *GcnT[2], *Wq[2], *Wk[2], *Wv[2], *Wo[2], *W1[2], *W2[2];
  for (int l = 0; l < 2; ++l) {
    GcnT[l] = (_Float16*)alloc((size_t)256 * 256 * 2);
    Wq[l]   = (_Float16*)alloc((size_t)256 * 256 * 2);
    Wk[l]   = (_Float16*)alloc((size_t)256 * 256 * 2);
    Wv[l]   = (_Float16*)alloc((size_t)256 * 256 * 2);
    Wo[l]   = (_Float16*)alloc((size_t)256 * 256 * 2);
    W1[l]   = (_Float16*)alloc((size_t)512 * 256 * 2);
    W2[l]   = (_Float16*)alloc((size_t)256 * 512 * 2);
  }
  float* deg  = (float*)alloc((size_t)NNODES * 4);
  float* dinv = (float*)alloc((size_t)NNODES * 4);
  float* BUF_H    = (float*)alloc(NM * 4);                 // node features
  float* BUF_HW   = (float*)alloc(NM * 4);                 // hw / attn O
  float* BUF_AGG  = (float*)alloc(NM * 4);                 // agg / pre-LN resid
  float* BUF_FLAT = (float*)alloc(NM * 4);                 // post-LN0
  float* BUF_F    = (float*)alloc((size_t)NNODES * FFND * 4);  // ffn hidden
  // QKV (f16) overlays BUF_F (disjoint lifetimes)
  _Float16* Q16 = (_Float16*)BUF_F;
  _Float16* K16 = Q16 + NM;
  _Float16* V16 = K16 + NM;

  const size_t GRU_SMEM = (size_t)(2 * 16 * GLD + 2 * 16 * HLD) * sizeof(float);
  const size_t ATT_SMEM = (size_t)128 * SLD * 4 + (size_t)128 * ALD * 2 + (size_t)32 * ALD * 2;
  hipFuncSetAttribute((const void*)gru_kernel,
                      hipFuncAttributeMaxDynamicSharedMemorySize, (int)GRU_SMEM);
  hipFuncSetAttribute((const void*)attn_kernel,
                      hipFuncAttributeMaxDynamicSharedMemorySize, (int)ATT_SMEM);

  // --- weight conversion to f16 (gcn_w transposed to Wt convention) ---
  auto cvt = [&](const float* s, _Float16* d, int r, int c, int t) {
    const int n = r * c;
    cvt_kernel<<<(n + 255) / 256, 256, 0, stream>>>(s, d, r, c, t);
  };
  cvt(f_wih0, Wih0, 768, 128, 0);
  cvt(f_whh0, Whh0, 768, 256, 0);
  cvt(f_wih1, Wih1, 768, 256, 0);
  cvt(f_whh1, Whh1, 768, 256, 0);
  for (int l = 0; l < 2; ++l) {
    cvt(P(l, 0),  GcnT[l], 256, 256, 1);   // h @ w  -> transpose
    cvt(P(l, 2),  Wq[l],   256, 256, 0);
    cvt(P(l, 4),  Wk[l],   256, 256, 0);
    cvt(P(l, 6),  Wv[l],   256, 256, 0);
    cvt(P(l, 8),  Wo[l],   256, 256, 0);
    cvt(P(l, 12), W1[l],   512, 256, 0);
    cvt(P(l, 14), W2[l],   256, 512, 0);
  }

  // --- fused 2-layer GRU -> BUF_H ---
  gru_kernel<<<NNODES / 16, 256, GRU_SMEM, stream>>>(
      x, Wih0, Whh0, Wih1, Whh1, b_ih0, b_hh0, b_ih1, b_hh1, BUF_H);

  // --- GCN degree normalization (shared by both layers) ---
  deg_init_kernel<<<NNODES / 256, 256, 0, stream>>>(deg);
  deg_edge_kernel<<<EDGES / 256, 256, 0, stream>>>(ei, deg);
  dinv_kernel<<<NNODES / 256, 256, 0, stream>>>(deg, dinv);

  for (int l = 0; l < 2; ++l) {
    // hw = h @ gcn_w
    launch_gemm(0, BUF_H, 256, GcnT[l], 256, 256, nullptr, nullptr, 0,
                BUF_HW, nullptr, 256, stream);
    // agg = bias + self-loop + scatter(msg)
    agg_init_kernel<<<(int)(NM / 256), 256, 0, stream>>>(BUF_HW, dinv, P(l, 1), BUF_AGG);
    gcn_edge_kernel<<<EDGES * 64 / 256, 256, 0, stream>>>(ei, BUF_HW, dinv, BUF_AGG);
    // h = relu(agg)
    relu_kernel<<<(int)(NM / 256), 256, 0, stream>>>(BUF_AGG, BUF_H);

    // Q,K,V (f16)
    launch_gemm(9, BUF_H, 256, Wq[l], 256, 256, P(l, 3), nullptr, 0, nullptr, Q16, 256, stream);
    launch_gemm(9, BUF_H, 256, Wk[l], 256, 256, P(l, 5), nullptr, 0, nullptr, K16, 256, stream);
    launch_gemm(9, BUF_H, 256, Wv[l], 256, 256, P(l, 7), nullptr, 0, nullptr, V16, 256, stream);
    // attention -> BUF_HW (as O)
    attn_kernel<<<NGRAPH * 8, 256, ATT_SMEM, stream>>>(Q16, K16, V16, BUF_HW);
    // o @ wo^T + bo + h  -> BUF_AGG ; LN0 -> BUF_FLAT
    launch_gemm(5, BUF_HW, 256, Wo[l], 256, 256, P(l, 9), BUF_H, 256,
                BUF_AGG, nullptr, 256, stream);
    ln_kernel<<<NNODES / 8, 256, 0, stream>>>(BUF_AGG, P(l, 10), P(l, 11), BUF_FLAT);
    // FFN
    launch_gemm(3, BUF_FLAT, 256, W1[l], 256, 512, P(l, 13), nullptr, 0,
                BUF_F, nullptr, 512, stream);
    launch_gemm(5, BUF_F, 512, W2[l], 512, 256, P(l, 15), BUF_FLAT, 256,
                BUF_AGG, nullptr, 256, stream);
    ln_kernel<<<NNODES / 8, 256, 0, stream>>>(BUF_AGG, P(l, 16), P(l, 17), BUF_H);
  }

  // --- global mean pool ---
  pool_kernel<<<NGRAPH, 256, 0, stream>>>(BUF_H, (float*)d_out);
}
